// MatchTensor_14173392076877
// MI455X (gfx1250) — matrix operand order repaired and checked by reading, not compiled, not run on hardware
//
#include <hip/hip_runtime.h>
#include <hip/hip_bf16.h>
#include <math.h>

// ---------------------------------------------------------------------------
// MatchTensor on MI455X (gfx1250): all contractions via v_wmma_f32_16x16x32_f16
// Fragment-native operand layouts + software-pipelined inner loops.
// ---------------------------------------------------------------------------

typedef __attribute__((ext_vector_type(16))) _Float16 v16h;
typedef __attribute__((ext_vector_type(8)))  float    v8f;

#define WMMA16x16x32(a, b, c) \
  __builtin_amdgcn_wmma_f32_16x16x32_f16(false, (a), false, (b), (short)0, (c), false, false)

// Problem dimensions (match reference)
static constexpr int BSZ  = 32;
static constexpr int DNUM = 10;
static constexpr int QL   = 16;
static constexpr int DLN  = 128;
static constexpr int BD   = BSZ * DNUM;          // 320
static constexpr int EMB  = 300;
static constexpr int FPAD = 320;                 // K padded to mult of 32
static constexpr int FN   = 304;                 // N padded to mult of 16 (valid 300)
static constexpr int HID  = 256;
static constexpr int G4   = 1024;
static constexpr int CCH  = 50;
static constexpr int CP   = 64;                  // conv input channels padded 51->64
static constexpr int NPIX = QL * DLN;            // 2048
static constexpr int NFLT = 50;
static constexpr int MFLT = 20;
static constexpr int FCH  = 160;                 // feature channels padded 150->160

// ---------------------------------------------------------------------------
// A fragment: 16x32 (MxK), row-major source, lda multiple of 8.
// lanes 0-15: row m, halves = K[0..7],K[16..23]; lanes 16-31: K[8..15],K[24..31]
__device__ inline v16h load_a16(const _Float16* __restrict__ A, int lda,
                                int m0, int k0, int lane) {
  int m  = m0 + (lane & 15);
  int kb = k0 + ((lane & 16) ? 8 : 0);
  const _Float16* p = A + (size_t)m * lda + kb;
  v16h r;
#pragma unroll
  for (int i = 0; i < 8; ++i) { r[i] = p[i]; r[8 + i] = p[16 + i]; }
  return r;
}

// B fragment from packed layout [ktile][ntile][lane][16 halves]:
// one aligned 32-byte load per lane.
__device__ inline v16h load_b16p(const _Float16* __restrict__ Bp, int Ntiles,
                                 int kt, int nt, int lane) {
  return *(const v16h*)(Bp + ((((size_t)kt * Ntiles + nt) * 32 + lane) << 4));
}

// C/D: element r of v8f <-> (m = m0 + r + 8*(lane>=16), n = n0 + (lane&15))

// ---------------------------------------------------------------------------
// Generic GEMM: out = A(f16 MxK row-major) @ Bp(packed) + bias0 + bias1
// Software-pipelined: next-iteration fragments issue before current WMMA.
__global__ __launch_bounds__(256)
void k_gemm(const _Float16* __restrict__ A, int lda,
            const _Float16* __restrict__ Bp,
            const float* __restrict__ bias0, const float* __restrict__ bias1,
            float* __restrict__ outF, _Float16* __restrict__ outH, int ldc,
            int Mtiles, int Ntiles, int K, int Nvalid) {
  int lane = threadIdx.x & 31;
  int wave = threadIdx.x >> 5;
  int tile = blockIdx.x * 8 + wave;
  if (tile >= Mtiles * Ntiles) return;           // wave-uniform exit
  int mt = tile / Ntiles, nt = tile - mt * Ntiles;
  int m0 = mt * 16, n0 = nt * 16;
  v8f c = {};
  v16h a = load_a16(A, lda, m0, 0, lane);
  v16h b = load_b16p(Bp, Ntiles, 0, nt, lane);
  for (int k0 = 32; k0 < K; k0 += 32) {
    __builtin_prefetch(A + (size_t)(m0 + (lane & 15)) * lda + k0 + 32, 0, 1);
    v16h an = load_a16(A, lda, m0, k0, lane);
    v16h bn = load_b16p(Bp, Ntiles, k0 >> 5, nt, lane);
    c = WMMA16x16x32(a, b, c);
    a = an; b = bn;
  }
  c = WMMA16x16x32(a, b, c);
  int n = n0 + (lane & 15);
  if (n < Nvalid) {
    float bv = (bias0 ? bias0[n] : 0.0f) + (bias1 ? bias1[n] : 0.0f);
    int mb = m0 + ((lane & 16) ? 8 : 0);
#pragma unroll
    for (int r = 0; r < 8; ++r) {
      float v = c[r] + bv;
      size_t off = (size_t)(mb + r) * ldc + n;
      if (outF) outF[off] = v;
      if (outH) outH[off] = (_Float16)v;
    }
  }
}

// ---------------------------------------------------------------------------
// Implicit-GEMM conv, HWC input: X [img][16][128][64] f16 (ch 51..63 zero).
// K-index = (ky*KW + kx)*64 + c  ->  each 8-half chunk is one contiguous load.
// Output: relu(conv+bias) -> F [img][2048][160] f16 at channel offset chOff.
template <int KW>
__device__ inline void conv_load8(v16h& a, int o, const _Float16* __restrict__ x,
                                  int qy, int dx, int kk) {
  int cell = kk >> 6;
  int coff = kk & 63;
  int ky = cell / KW;
  int kx = cell - ky * KW;
  int y  = qy + ky - 1;
  int xx = dx + kx - (KW / 2);
  if (y >= 0 && y < QL && xx >= 0 && xx < DLN) {
    const _Float16* p = x + (((size_t)y * DLN + xx) << 6) + coff;
#pragma unroll
    for (int i = 0; i < 8; ++i) a[o + i] = p[i];
  } else {
#pragma unroll
    for (int i = 0; i < 8; ++i) a[o + i] = (_Float16)0.0f;
  }
}

template <int KW>
__device__ inline v16h conv_frag(const _Float16* __restrict__ x,
                                 int qy, int dx, int k0, int lane) {
  int kb = k0 + ((lane & 16) ? 8 : 0);
  v16h a;
  conv_load8<KW>(a, 0, x, qy, dx, kb);
  conv_load8<KW>(a, 8, x, qy, dx, kb + 16);
  return a;
}

template <int KW>
__global__ __launch_bounds__(256)
void k_conv(const _Float16* __restrict__ X, const _Float16* __restrict__ Wp,
            const float* __restrict__ bias, _Float16* __restrict__ Fout,
            int chOff) {
  constexpr int KTOT = 3 * KW * CP;              // 576 / 960 / 1344 (mult of 32)
  int lane = threadIdx.x & 31;
  int wave = threadIdx.x >> 5;
  int tile = blockIdx.x * 8 + wave;
  if (tile >= BD * 512) return;                  // 128 m-tiles * 4 n-tiles / image
  int img = tile >> 9;
  int t   = tile & 511;
  int mt  = t >> 2, nt = t & 3;
  const _Float16* x = X + (size_t)img * NPIX * CP;
  int m0 = mt * 16, n0 = nt * 16;
  int m  = m0 + (lane & 15);
  int qy = m >> 7, dx = m & 127;
  v8f c = {};
  v16h a = conv_frag<KW>(x, qy, dx, 0, lane);
  v16h b = load_b16p(Wp, 4, 0, nt, lane);
  for (int k0 = 32; k0 < KTOT; k0 += 32) {
    v16h an = conv_frag<KW>(x, qy, dx, k0, lane);
    v16h bn = load_b16p(Wp, 4, k0 >> 5, nt, lane);
    c = WMMA16x16x32(a, b, c);
    a = an; b = bn;
  }
  c = WMMA16x16x32(a, b, c);
  int n = n0 + (lane & 15);
  if (n < NFLT) {
    float bv = bias[n];
    int mb = m0 + ((lane & 16) ? 8 : 0);
#pragma unroll
    for (int r = 0; r < 8; ++r) {
      float v = c[r] + bv;
      v = v > 0.0f ? v : 0.0f;
      Fout[((size_t)img * NPIX + (mb + r)) * FCH + chOff + n] = (_Float16)v;
    }
  }
}

// ---------------------------------------------------------------------------
// Fused 1x1 conv (150->20) + global max pool.  F [img][2048][160] f16 (A path
// is plain contiguous load_a16), W packed [5][2][lane][16].
__device__ inline void atomicMaxF(float* addr, float val) {
  unsigned int* ua = (unsigned int*)addr;
  unsigned int old = *ua;
  while (__uint_as_float(old) < val) {
    unsigned int assumed = old;
    old = atomicCAS(ua, assumed, __float_as_uint(val));
    if (old == assumed) break;
  }
}

__global__ __launch_bounds__(256)
void k_cc_pool(const _Float16* __restrict__ F, const _Float16* __restrict__ Wp,
               const float* __restrict__ bias, float* __restrict__ pooled) {
  int lane = threadIdx.x & 31;
  int wave = threadIdx.x >> 5;
  int tile = blockIdx.x * 8 + wave;
  if (tile >= BD * 256) return;                  // 128 m-tiles * 2 n-tiles / image
  int img = tile >> 8;
  int t   = tile & 255;
  int mt  = t >> 1, nt = t & 1;
  const _Float16* Fi = F + (size_t)img * NPIX * FCH;
  int m0 = mt * 16, n0 = nt * 16;
  v8f c = {};
  v16h a = load_a16(Fi, FCH, m0, 0, lane);
  v16h b = load_b16p(Wp, 2, 0, nt, lane);
  for (int k0 = 32; k0 < FCH; k0 += 32) {
    v16h an = load_a16(Fi, FCH, m0, k0, lane);
    v16h bn = load_b16p(Wp, 2, k0 >> 5, nt, lane);
    c = WMMA16x16x32(a, b, c);
    a = an; b = bn;
  }
  c = WMMA16x16x32(a, b, c);
  int n = n0 + (lane & 15);
  float mx = -3.0e38f;
  if (n < MFLT) {
    float bv = bias[n];
#pragma unroll
    for (int r = 0; r < 8; ++r) mx = fmaxf(mx, c[r] + bv);
  }
  float other = __shfl_xor(mx, 16, 32);          // combine row halves (wave32)
  mx = fmaxf(mx, other);
  if (lane < 16 && n < MFLT) atomicMaxF(&pooled[img * MFLT + n], mx);
}

// ---------------------------------------------------------------------------
// Fills / packing / elementwise kernels
__global__ void k_fill_f32(float* p, float v, size_t n) {
  size_t i = (size_t)blockIdx.x * blockDim.x + threadIdx.x;
  if (i < n) p[i] = v;
}
__global__ void k_fill_f16(_Float16* p, float v, size_t n) {
  size_t i = (size_t)blockIdx.x * blockDim.x + threadIdx.x;
  if (i < n) p[i] = (_Float16)v;
}

// Decode packed-B destination index -> (k, n)
__device__ inline void bpack_decode(int i, int Ntiles, int& k, int& n) {
  int hi   = i & 15;
  int lane = (i >> 4) & 31;
  int tile = i >> 9;
  int nt = tile % Ntiles, kt = tile / Ntiles;
  int kk = (hi & 7) + ((lane >> 4) << 3) + ((hi >> 3) << 4);
  k = kt * 32 + kk;
  n = nt * 16 + (lane & 15);
}
// src KxN row-major -> packed
__global__ void k_pack_kn(const float* __restrict__ src, _Float16* __restrict__ dst,
                          int K0, int N0, int Kpad, int Ntiles) {
  int i = blockIdx.x * blockDim.x + threadIdx.x;
  if (i >= Kpad * Ntiles * 16) return;
  int k, n; bpack_decode(i, Ntiles, k, n);
  float v = (k < K0 && n < N0) ? src[(size_t)k * N0 + n] : 0.0f;
  dst[i] = (_Float16)v;
}
// src NxK row-major (transpose) -> packed
__global__ void k_pack_nk(const float* __restrict__ src, _Float16* __restrict__ dst,
                          int K0, int N0, int Kpad, int Ntiles) {
  int i = blockIdx.x * blockDim.x + threadIdx.x;
  if (i >= Kpad * Ntiles * 16) return;
  int k, n; bpack_decode(i, Ntiles, k, n);
  float v = (k < K0 && n < N0) ? src[(size_t)n * K0 + k] : 0.0f;
  dst[i] = (_Float16)v;
}
// conv weight [50][51][3][kw] -> packed with k = (ky*kw+kx)*64 + c, Ntiles=4
__global__ void k_pack_convw(const float* __restrict__ src, _Float16* __restrict__ dst,
                             int kw, int Ktot) {
  int i = blockIdx.x * blockDim.x + threadIdx.x;
  if (i >= Ktot * 64) return;
  int k, n; bpack_decode(i, 4, k, n);
  int cell = k >> 6, c = k & 63;
  int ky = cell / kw, kx = cell - ky * kw;
  _Float16 v = (_Float16)0.0f;
  if (n < NFLT && c < 51) {
    v = (_Float16)src[(((size_t)n * 51 + c) * 3 + ky) * kw + kx];
  }
  dst[i] = v;
}
// gather embedding rows -> f16, K padded 300->320 with zeros
__global__ void k_gather_embed(const int* __restrict__ idx,
                               const float* __restrict__ emb,
                               _Float16* __restrict__ out, int nrows) {
  long long i = (long long)blockIdx.x * blockDim.x + threadIdx.x;
  if (i >= (long long)nrows * FPAD) return;
  int r = (int)(i / FPAD), k = (int)(i - (long long)r * FPAD);
  _Float16 v = (_Float16)0.0f;
  if (k < EMB) v = (_Float16)emb[(size_t)idx[r] * EMB + k];
  out[i] = v;
}

__device__ inline float sigmoidf(float x) { return 1.0f / (1.0f + expf(-x)); }

// LSTM gate update for one timestep.
__global__ void k_lstm_gates(const float* __restrict__ xg,
                             const float* __restrict__ glin,
                             float* __restrict__ cst, _Float16* __restrict__ h16,
                             _Float16* __restrict__ enc, int Nseq, int T, int t) {
  int i = blockIdx.x * blockDim.x + threadIdx.x;
  if (i >= Nseq * HID) return;
  int seq = i >> 8, hh = i & 255;
  size_t xrow = ((size_t)seq * T + t) * G4;
  size_t grow = (size_t)seq * G4;
  float gi = xg[xrow + hh]           + glin[grow + hh];
  float gf = xg[xrow + HID + hh]     + glin[grow + HID + hh];
  float gg = xg[xrow + 2 * HID + hh] + glin[grow + 2 * HID + hh];
  float go = xg[xrow + 3 * HID + hh] + glin[grow + 3 * HID + hh];
  float cv = sigmoidf(gf) * cst[i] + sigmoidf(gi) * tanhf(gg);
  float hv = sigmoidf(go) * tanhf(cv);
  cst[i] = cv;
  h16[i] = (_Float16)hv;
  enc[((size_t)seq * T + t) * HID + hh] = (_Float16)hv;
}

// Build match tensor X f16 HWC [img][16][128][64]:
//   c<50: pq[b,q,c]*pd[img,d,c]; c==50: (qtok==dtok)*alpha; c>50: 0
__global__ void k_build_x(const float* __restrict__ pq, const float* __restrict__ pd,
                          const int* __restrict__ bq, const int* __restrict__ bdocs,
                          const float* __restrict__ alpha, _Float16* __restrict__ X) {
  long long i = (long long)blockIdx.x * blockDim.x + threadIdx.x;
  const long long total = (long long)BD * NPIX * CP;
  if (i >= total) return;
  int cch = (int)(i & 63);
  int m   = (int)((i >> 6) & (NPIX - 1));
  int img = (int)(i >> 17);                      // NPIX*CP = 2^17
  int q = m >> 7, d = m & 127;
  int b = img / DNUM, dn = img - b * DNUM;
  float v = 0.0f;
  if (cch < CCH) {
    v = pq[((size_t)(b * QL + q)) * 64 + cch] * pd[((size_t)(img * DLN + d)) * 64 + cch];
  } else if (cch == CCH) {
    int qt = bq[b * QL + q];
    int dt = bdocs[(b * DNUM + dn) * DLN + d];
    v = (qt == dt) ? alpha[0] : 0.0f;
  }
  X[i] = (_Float16)v;
}

__global__ void k_final(const float* __restrict__ pooled, const float* __restrict__ outW,
                        const float* __restrict__ outB, float* __restrict__ out) {
  int i = blockIdx.x * blockDim.x + threadIdx.x;
  if (i >= BD) return;
  float s = outB[0];
#pragma unroll
  for (int f = 0; f < MFLT; ++f) s += pooled[i * MFLT + f] * outW[f];
  out[i] = s;
}

// ---------------------------------------------------------------------------
static void launch_gemm(const _Float16* A, int lda, const _Float16* Bp,
                        const float* b0, const float* b1, float* oF, _Float16* oH,
                        int ldc, int Mt, int Nt, int K, int Nvalid, hipStream_t s) {
  int blocks = (Mt * Nt + 7) / 8;
  k_gemm<<<blocks, 256, 0, s>>>(A, lda, Bp, b0, b1, oF, oH, ldc, Mt, Nt, K, Nvalid);
}
static inline int cdiv(long long n, int b) { return (int)((n + b - 1) / b); }

extern "C" void kernel_launch(void* const* d_in, const int* in_sizes, int n_in,
                              void* d_out, int out_size, void* d_ws, size_t ws_size,
                              hipStream_t stream) {
  (void)in_sizes; (void)n_in; (void)out_size; (void)ws_size;
  const int*   bq     = (const int*)d_in[0];
  const int*   bdocs  = (const int*)d_in[1];
  const float* emb    = (const float*)d_in[4];
  const float* proj_W = (const float*)d_in[5];
  const float* proj_b = (const float*)d_in[6];
  const float* q_Wih  = (const float*)d_in[7];
  const float* q_Whh  = (const float*)d_in[8];
  const float* q_bih  = (const float*)d_in[9];
  const float* q_bhh  = (const float*)d_in[10];
  const float* d_Wih  = (const float*)d_in[11];
  const float* d_Whh  = (const float*)d_in[12];
  const float* d_bih  = (const float*)d_in[13];
  const float* d_bhh  = (const float*)d_in[14];
  const float* qp_W   = (const float*)d_in[15];
  const float* qp_b   = (const float*)d_in[16];
  const float* dp_W   = (const float*)d_in[17];
  const float* dp_b   = (const float*)d_in[18];
  const float* alpha  = (const float*)d_in[19];
  const float* c1_W = (const float*)d_in[20]; const float* c1_b = (const float*)d_in[21];
  const float* c2_W = (const float*)d_in[22]; const float* c2_b = (const float*)d_in[23];
  const float* c3_W = (const float*)d_in[24]; const float* c3_b = (const float*)d_in[25];
  const float* cc_W = (const float*)d_in[26]; const float* cc_b = (const float*)d_in[27];
  const float* out_W = (const float*)d_in[28]; const float* out_b = (const float*)d_in[29];
  float* out = (float*)d_out;

  // Workspace carve-out (256B aligned)
  char* wp = (char*)d_ws;
  auto alloc = [&](size_t bytes) -> char* {
    char* r = wp; wp += (bytes + 255) & ~(size_t)255; return r;
  };
  _Float16* Wp16   = (_Float16*)alloc((size_t)FPAD * FN * 2);
  _Float16* qWih16 = (_Float16*)alloc((size_t)FPAD * G4 * 2);
  _Float16* dWih16 = (_Float16*)alloc((size_t)FPAD * G4 * 2);
  _Float16* qWhh16 = (_Float16*)alloc((size_t)HID * G4 * 2);
  _Float16* dWhh16 = (_Float16*)alloc((size_t)HID * G4 * 2);
  _Float16* qpW16  = (_Float16*)alloc((size_t)HID * 64 * 2);
  _Float16* dpW16  = (_Float16*)alloc((size_t)HID * 64 * 2);
  _Float16* w1_16  = (_Float16*)alloc((size_t)576 * 64 * 2);
  _Float16* w2_16  = (_Float16*)alloc((size_t)960 * 64 * 2);
  _Float16* w3_16  = (_Float16*)alloc((size_t)1344 * 64 * 2);
  _Float16* ccW16  = (_Float16*)alloc((size_t)FCH * 32 * 2);
  _Float16* Aq16   = (_Float16*)alloc((size_t)512 * FPAD * 2);
  _Float16* Ad16   = (_Float16*)alloc((size_t)40960 * FPAD * 2);
  _Float16* eq16   = (_Float16*)alloc((size_t)512 * FPAD * 2);
  _Float16* ed16   = (_Float16*)alloc((size_t)40960 * FPAD * 2);
  float*    xgq    = (float*)alloc((size_t)512 * G4 * 4);
  float*    xgd    = (float*)alloc((size_t)40960 * G4 * 4);
  _Float16* hq16   = (_Float16*)alloc((size_t)32 * HID * 2);
  float*    cq     = (float*)alloc((size_t)32 * HID * 4);
  _Float16* hd16   = (_Float16*)alloc((size_t)BD * HID * 2);
  float*    cd     = (float*)alloc((size_t)BD * HID * 4);
  float*    glq    = (float*)alloc((size_t)32 * G4 * 4);
  float*    gld    = (float*)alloc((size_t)BD * G4 * 4);
  _Float16* encq   = (_Float16*)alloc((size_t)512 * HID * 2);
  _Float16* encd   = (_Float16*)alloc((size_t)40960 * HID * 2);
  float*    pqb    = (float*)alloc((size_t)512 * 64 * 4);
  float*    pdb    = (float*)alloc((size_t)40960 * 64 * 4);
  _Float16* x16    = (_Float16*)alloc((size_t)BD * NPIX * CP * 2);
  _Float16* fbuf   = (_Float16*)alloc((size_t)BD * NPIX * FCH * 2);
  float*    pooled = (float*)alloc((size_t)BD * MFLT * 4);

  // --- init ---
  k_fill_f16<<<cdiv((long long)512 * FPAD, 256), 256, 0, stream>>>(eq16, 0.0f, (size_t)512 * FPAD);
  k_fill_f16<<<cdiv((long long)40960 * FPAD, 256), 256, 0, stream>>>(ed16, 0.0f, (size_t)40960 * FPAD);
  k_fill_f16<<<cdiv(32 * HID, 256), 256, 0, stream>>>(hq16, 0.0f, (size_t)32 * HID);
  k_fill_f16<<<cdiv(BD * HID, 256), 256, 0, stream>>>(hd16, 0.0f, (size_t)BD * HID);
  k_fill_f32<<<cdiv(32 * HID, 256), 256, 0, stream>>>(cq, 0.0f, (size_t)32 * HID);
  k_fill_f32<<<cdiv(BD * HID, 256), 256, 0, stream>>>(cd, 0.0f, (size_t)BD * HID);
  k_fill_f32<<<cdiv(BD * MFLT, 256), 256, 0, stream>>>(pooled, -3.0e38f, (size_t)BD * MFLT);
  k_fill_f16<<<cdiv((long long)BD * NPIX * FCH, 256), 256, 0, stream>>>(
      fbuf, 0.0f, (size_t)BD * NPIX * FCH);   // zero pad channels 150..159

  // --- weight packing (fragment-native B layout) ---
  k_pack_kn<<<cdiv(FPAD * FN, 256), 256, 0, stream>>>(proj_W, Wp16, EMB, EMB, FPAD, FN / 16);
  k_pack_nk<<<cdiv(FPAD * G4, 256), 256, 0, stream>>>(q_Wih, qWih16, EMB, G4, FPAD, 64);
  k_pack_nk<<<cdiv(FPAD * G4, 256), 256, 0, stream>>>(d_Wih, dWih16, EMB, G4, FPAD, 64);
  k_pack_nk<<<cdiv(HID * G4, 256), 256, 0, stream>>>(q_Whh, qWhh16, HID, G4, HID, 64);
  k_pack_nk<<<cdiv(HID * G4, 256), 256, 0, stream>>>(d_Whh, dWhh16, HID, G4, HID, 64);
  k_pack_kn<<<cdiv(HID * 64, 256), 256, 0, stream>>>(qp_W, qpW16, HID, CCH, HID, 4);
  k_pack_kn<<<cdiv(HID * 64, 256), 256, 0, stream>>>(dp_W, dpW16, HID, CCH, HID, 4);
  k_pack_nk<<<cdiv(FCH * 32, 256), 256, 0, stream>>>(cc_W, ccW16, 150, MFLT, FCH, 2);
  k_pack_convw<<<cdiv(576 * 64, 256), 256, 0, stream>>>(c1_W, w1_16, 3, 576);
  k_pack_convw<<<cdiv(960 * 64, 256), 256, 0, stream>>>(c2_W, w2_16, 5, 960);
  k_pack_convw<<<cdiv(1344 * 64, 256), 256, 0, stream>>>(c3_W, w3_16, 7, 1344);

  // --- embedding gather (f16, K-padded) ---
  k_gather_embed<<<cdiv((long long)512 * FPAD, 256), 256, 0, stream>>>(bq, emb, Aq16, 512);
  k_gather_embed<<<cdiv((long long)40960 * FPAD, 256), 256, 0, stream>>>(bdocs, emb, Ad16, 40960);

  // --- projection GEMMs: eq/ed = gathered @ proj_W + proj_b (f16 out) ---
  launch_gemm(Aq16, FPAD, Wp16, proj_b, nullptr, nullptr, eq16, FPAD,
              512 / 16, FN / 16, FPAD, EMB, stream);
  launch_gemm(Ad16, FPAD, Wp16, proj_b, nullptr, nullptr, ed16, FPAD,
              40960 / 16, FN / 16, FPAD, EMB, stream);

  // --- LSTM input precompute: xg = x @ Wih^T + bih + bhh (f32 out) ---
  launch_gemm(eq16, FPAD, qWih16, q_bih, q_bhh, xgq, nullptr, G4,
              512 / 16, 64, FPAD, G4, stream);
  launch_gemm(ed16, FPAD, dWih16, d_bih, d_bhh, xgd, nullptr, G4,
              40960 / 16, 64, FPAD, G4, stream);

  // --- query LSTM (32 seqs, 16 steps) ---
  for (int t = 0; t < QL; ++t) {
    launch_gemm(hq16, HID, qWhh16, nullptr, nullptr, glq, nullptr, G4,
                2, 64, HID, G4, stream);
    k_lstm_gates<<<cdiv(32 * HID, 256), 256, 0, stream>>>(xgq, glq, cq, hq16, encq, 32, QL, t);
  }
  // --- doc LSTM (320 seqs, 128 steps) ---
  for (int t = 0; t < DLN; ++t) {
    launch_gemm(hd16, HID, dWhh16, nullptr, nullptr, gld, nullptr, G4,
                BD / 16, 64, HID, G4, stream);
    k_lstm_gates<<<cdiv(BD * HID, 256), 256, 0, stream>>>(xgd, gld, cd, hd16, encd, BD, DLN, t);
  }

  // --- channel projections pq/pd (f32, ldc=64, valid 50) ---
  launch_gemm(encq, HID, qpW16, qp_b, nullptr, pqb, nullptr, 64,
              512 / 16, 4, HID, CCH, stream);
  launch_gemm(encd, HID, dpW16, dp_b, nullptr, pdb, nullptr, 64,
              40960 / 16, 4, HID, CCH, stream);

  // --- match tensor (HWC, padded channels) ---
  k_build_x<<<cdiv((long long)BD * NPIX * CP, 256), 256, 0, stream>>>(
      pqb, pdb, bq, bdocs, alpha, x16);

  // --- convs (implicit GEMM, relu, f16 out, pixel-major features) ---
  int convBlocks = BD * 512 / 8;
  k_conv<3><<<convBlocks, 256, 0, stream>>>(x16, w1_16, c1_b, fbuf, 0);
  k_conv<5><<<convBlocks, 256, 0, stream>>>(x16, w2_16, c2_b, fbuf, 50);
  k_conv<7><<<convBlocks, 256, 0, stream>>>(x16, w3_16, c3_b, fbuf, 100);

  // --- 1x1 conv + max pool ---
  k_cc_pool<<<BD * 256 / 8, 256, 0, stream>>>(fbuf, ccW16, cc_b, pooled);

  // --- final scores [32,10] ---
  k_final<<<cdiv(BD, 256), 256, 0, stream>>>(pooled, out_W, out_b, out);
}
